// TransformerModel_63307817943234
// MI455X (gfx1250) — matrix operand order repaired and checked by reading, not compiled, not run on hardware
//
#include <hip/hip_runtime.h>
#include <hip/hip_bf16.h>
#include <stdint.h>

// ---------------- model constants ----------------
constexpr int  B_  = 256;
constexpr int  T_  = 256;
constexpr int  D_  = 384;
constexpr int  H_  = 6;
constexpr int  HD_ = 64;
constexpr int  L_  = 6;
constexpr int  V_  = 65;
constexpr int  DF_ = 1536;           // 4*D
constexpr long BT_ = (long)B_ * T_;  // 65536

typedef __bf16 bf16;
typedef __bf16 v8bf  __attribute__((ext_vector_type(8)));
typedef __bf16 v16bf __attribute__((ext_vector_type(16)));
typedef float  v8f   __attribute__((ext_vector_type(8)));

#define WMMA_BF16(a, b, c) \
  __builtin_amdgcn_wmma_f32_16x16x32_bf16(false, (a), false, (b), (short)0, (c), false, false)

static __device__ __forceinline__ v16bf combine8(v8bf lo, v8bf hi) {
  return __builtin_shufflevector(lo, hi, 0,1,2,3,4,5,6,7,8,9,10,11,12,13,14,15);
}

// ---------------- fp32 [K][N] -> bf16 [N][K] transpose-convert (weights, once) ----
__global__ void k_cvtT(const float* __restrict__ s, bf16* __restrict__ d,
                       int K, int N, int Lb) {
  const long kn = (long)K * N;
  const long total = kn * Lb;
  long i = (long)blockIdx.x * blockDim.x + threadIdx.x;
  const long stride = (long)gridDim.x * blockDim.x;
  for (; i < total; i += stride) {
    long lyr = i / kn;
    long r   = i - lyr * kn;
    int  k   = (int)(r / N);
    int  n   = (int)(r - (long)k * N);
    d[lyr * kn + (size_t)n * K + k] = (bf16)s[i];
  }
}

// ---------------- embedding: x = tok_emb[idx] + pos_emb ----------------
__global__ __launch_bounds__(128) void k_embed(const int* __restrict__ idx,
                                               const float* __restrict__ tok,
                                               const float* __restrict__ pos,
                                               float* __restrict__ x,
                                               bf16* __restrict__ xb) {
  const long t  = blockIdx.x;          // token 0..BT-1
  const int  tt = (int)(t % T_);
  const int  v  = idx[t];
  const float* tr = tok + (size_t)v * D_;
  const float* pr = pos + (size_t)tt * D_;
  float* xr  = x  + (size_t)t * D_;
  bf16*  xbr = xb + (size_t)t * D_;
  for (int d = threadIdx.x; d < D_; d += 128) {
    float val = tr[d] + pr[d];
    xr[d]  = val;
    xbr[d] = (bf16)val;
  }
}

// ---------------- tiled bf16 WMMA GEMM (double-buffered LDS) ----------------
// C[M,N] = A[M,K] (bf16, row-major) * W[K,N], with W given TRANSPOSED: Wt[N][K] bf16.
// flags: bit0 = add bias[n], bit1 = relu.  Cf / Cb may each be null.
// Requires: M % 128 == 0, K % 32 == 0.  N edge handled by row-clamp + store guard.
#define BM 128
#define BN 128
#define BK 32
__global__ __launch_bounds__(256) void k_gemm(const bf16* __restrict__ A,
                                              const bf16* __restrict__ Wt,
                                              const float* __restrict__ bias,
                                              float* Cf, bf16* Cb,
                                              int M, int N, int K, int flags) {
  __shared__ __align__(32) bf16 As[2 * BM * BK];
  __shared__ __align__(32) bf16 Bs[2 * BN * BK];

  const int tid = threadIdx.x;
  const int l   = tid & 31;
  const int wid = tid >> 5;
  const int nl  = l & 15;
  const int hl  = l >> 4;
  const int waveM = wid & 3;   // 4 waves along M  (4*32 = 128)
  const int waveN = wid >> 2;  // 2 waves along N  (2*64 = 128)
  const int m0 = blockIdx.y * BM;
  const int n0 = blockIdx.x * BN;

  // staging: each thread covers (row, col..col+7) and (row+64, col..col+7)
  const int row = tid >> 2;           // 0..63
  const int col = (tid & 3) << 3;     // 0,8,16,24
  // clamp B rows: output column n only depends on Wt row n; cols >= N are never stored
  int gn0 = n0 + row;      if (gn0 > N - 1) gn0 = N - 1;
  int gn1 = n0 + row + 64; if (gn1 > N - 1) gn1 = N - 1;

  const bf16* pa0 = A  + (size_t)(m0 + row) * K + col;
  const bf16* pa1 = pa0 + (size_t)64 * K;
  const bf16* pb0 = Wt + (size_t)gn0 * K + col;
  const bf16* pb1 = Wt + (size_t)gn1 * K + col;
  const int s0 = row * BK + col;            // LDS staging offsets (halves)
  const int s1 = (row + 64) * BK + col;

  // stage K-tile 0 into buffer 0
  uint4 a0 = *(const uint4*)pa0;
  uint4 a1 = *(const uint4*)pa1;
  uint4 b0 = *(const uint4*)pb0;
  uint4 b1 = *(const uint4*)pb1;
  *(uint4*)&As[s0] = a0;
  *(uint4*)&As[s1] = a1;
  *(uint4*)&Bs[s0] = b0;
  *(uint4*)&Bs[s1] = b1;
  __syncthreads();

  v8f acc[2][4] = {};
  const int nkt = K / BK;
  for (int kt = 0; kt < nkt; ++kt) {
    const int cur  = (kt & 1) * (BM * BK);
    const bool more = (kt + 1) < nkt;
    if (more) {  // issue next tile's global loads before the WMMAs
      const int ko = (kt + 1) * BK;
      a0 = *(const uint4*)(pa0 + ko);
      a1 = *(const uint4*)(pa1 + ko);
      b0 = *(const uint4*)(pb0 + ko);
      b1 = *(const uint4*)(pb1 + ko);
    }

    const bf16* Abase = &As[cur];
    const bf16* Bbase = &Bs[cur];
    v16bf af[2], bfr[4];
#pragma unroll
    for (int mt = 0; mt < 2; ++mt) {
      const bf16* p = Abase + (waveM * 32 + mt * 16 + nl) * BK + hl * 16;
      af[mt] = *(const v16bf*)p;
    }
#pragma unroll
    for (int nt = 0; nt < 4; ++nt)
      bfr[nt] = *(const v16bf*)(Bbase + (waveN * 64 + nt * 16 + nl) * BK + hl * 16);
#pragma unroll
    for (int mt = 0; mt < 2; ++mt)
#pragma unroll
      for (int nt = 0; nt < 4; ++nt)
        acc[mt][nt] = WMMA_BF16(af[mt], bfr[nt], acc[mt][nt]);

    if (more) {  // store next tile into the other buffer, after compute
      const int nxt = ((kt + 1) & 1) * (BM * BK);
      *(uint4*)&As[nxt + s0] = a0;
      *(uint4*)&As[nxt + s1] = a1;
      *(uint4*)&Bs[nxt + s0] = b0;
      *(uint4*)&Bs[nxt + s1] = b1;
    }
    __syncthreads();
  }

  // epilogue
#pragma unroll
  for (int nt = 0; nt < 4; ++nt) {
    int n = n0 + waveN * 64 + nt * 16 + nl;
    if (n >= N) continue;
    float bb = (flags & 1) ? bias[n] : 0.f;
#pragma unroll
    for (int mt = 0; mt < 2; ++mt) {
#pragma unroll
      for (int i = 0; i < 8; ++i) {
        int m = m0 + waveM * 32 + mt * 16 + i + 8 * hl;
        float v = acc[mt][nt][i] + bb;
        if (flags & 2) v = fmaxf(v, 0.f);
        if (Cf) Cf[(size_t)m * N + n] = v;
        if (Cb) Cb[(size_t)m * N + n] = (bf16)v;
      }
    }
  }
}

// ---------------- flash attention, one wave per (b, h, 16-query tile) ----------------
__global__ __launch_bounds__(32) void k_attn(const bf16* __restrict__ Q,
                                             const bf16* __restrict__ Kmat,
                                             const bf16* __restrict__ Vmat,
                                             bf16* __restrict__ O) {
  __shared__ __align__(32) bf16 Pt[16 * 32];  // P tile, row-major [16 q][32 keys]
  __shared__ __align__(32) bf16 Vs[64 * 32];  // V block transposed [64 dims][32 keys]

  const int l  = threadIdx.x;
  const int nl = l & 15;
  const int hl = l >> 4;
  const int q0 = blockIdx.x * 16;
  const int h  = blockIdx.y;
  const int b  = blockIdx.z;
  const size_t base = (size_t)b * T_ * D_ + (size_t)h * HD_;

  // Q A-fragments: rows q0+nl, HD split into two K-chunks of 32
  v16bf qf[2];
  {
    const bf16* qr = Q + base + (size_t)(q0 + nl) * D_;
#pragma unroll
    for (int c = 0; c < 2; ++c) {
      const bf16* p = qr + c * 32 + hl * 16;
      qf[c] = *(const v16bf*)p;
    }
  }

  v8f out[4] = {};
  float mrun[8], lsum[8];
#pragma unroll
  for (int i = 0; i < 8; ++i) { mrun[i] = -1e30f; lsum[i] = 0.f; }

  const int nblk = (q0 + 15) / 32 + 1;  // causal: only key blocks <= query tile
  for (int jb = 0; jb < nblk; ++jb) {
    const int j = jb * 32;

    // S = Q @ K^T for two 16-key subtiles
    v8f sc[2] = {};
#pragma unroll
    for (int kt2 = 0; kt2 < 2; ++kt2) {
      const bf16* kr = Kmat + base + (size_t)(j + kt2 * 16 + nl) * D_;
#pragma unroll
      for (int c = 0; c < 2; ++c) {
        v16bf kf = *(const v16bf*)(kr + c * 32 + hl * 16);
        sc[kt2] = WMMA_BF16(qf[c], kf, sc[kt2]);
      }
    }

    // scale + causal mask + online softmax
    float p[2][8], rm[8];
#pragma unroll
    for (int i = 0; i < 8; ++i) {
      const int qi = q0 + i + 8 * hl;
      float a = sc[0][i] * 0.125f; if (j + nl      > qi) a = -1e30f;
      float c = sc[1][i] * 0.125f; if (j + 16 + nl > qi) c = -1e30f;
      p[0][i] = a; p[1][i] = c; rm[i] = fmaxf(a, c);
    }
#pragma unroll
    for (int o = 1; o < 16; o <<= 1)
#pragma unroll
      for (int i = 0; i < 8; ++i) rm[i] = fmaxf(rm[i], __shfl_xor(rm[i], o, 32));
    float alpha[8];
#pragma unroll
    for (int i = 0; i < 8; ++i) {
      float mn = fmaxf(mrun[i], rm[i]);
      alpha[i] = __expf(mrun[i] - mn);
      mrun[i]  = mn;
      p[0][i] = __expf(p[0][i] - mn);
      p[1][i] = __expf(p[1][i] - mn);
    }
    float rsum[8];
#pragma unroll
    for (int i = 0; i < 8; ++i) rsum[i] = p[0][i] + p[1][i];
#pragma unroll
    for (int o = 1; o < 16; o <<= 1)
#pragma unroll
      for (int i = 0; i < 8; ++i) rsum[i] += __shfl_xor(rsum[i], o, 32);
#pragma unroll
    for (int i = 0; i < 8; ++i) lsum[i] = lsum[i] * alpha[i] + rsum[i];
#pragma unroll
    for (int nd = 0; nd < 4; ++nd)
#pragma unroll
      for (int i = 0; i < 8; ++i) out[nd][i] *= alpha[i];

    // P tile -> LDS (C layout -> row-major), V block -> LDS transposed
#pragma unroll
    for (int kt2 = 0; kt2 < 2; ++kt2)
#pragma unroll
      for (int i = 0; i < 8; ++i)
        Pt[(i + 8 * hl) * 32 + kt2 * 16 + nl] = (bf16)p[kt2][i];
    {
      const bf16* vr = Vmat + base + (size_t)(j + l) * D_;   // one key row per lane
#pragma unroll
      for (int d0 = 0; d0 < HD_; d0 += 8) {
        v8bf vv = *(const v8bf*)(vr + d0);
#pragma unroll
        for (int i = 0; i < 8; ++i) Vs[(d0 + i) * 32 + l] = vv[i];
      }
    }
    asm volatile("s_wait_dscnt 0x0" ::: "memory");  // single wave: stores -> loads

    v16bf pf;
    {
      const bf16* pp = &Pt[nl * 32 + hl * 16];
      pf = *(const v16bf*)pp;
    }
#pragma unroll
    for (int nd = 0; nd < 4; ++nd) {
      v16bf vf = *(const v16bf*)&Vs[(nd * 16 + nl) * 32 + hl * 16];
      out[nd] = WMMA_BF16(pf, vf, out[nd]);
    }
    asm volatile("s_wait_dscnt 0x0" ::: "memory");  // keep WAR ordering vs next iter
  }

  // normalize and store (bf16, head-interleaved layout)
  float inv[8];
#pragma unroll
  for (int i = 0; i < 8; ++i) inv[i] = 1.f / lsum[i];
#pragma unroll
  for (int nd = 0; nd < 4; ++nd)
#pragma unroll
    for (int i = 0; i < 8; ++i)
      O[base + (size_t)(q0 + i + 8 * hl) * D_ + nd * 16 + nl] = (bf16)(out[nd][i] * inv[i]);
}

// ---------------- x = LayerNorm(x)*g + b + proj  (wave per row) ----------------
__global__ __launch_bounds__(256) void k_ln_residual(const float* x,
                                                     const float* __restrict__ g,
                                                     const float* __restrict__ be,
                                                     const float* __restrict__ proj,
                                                     float* xo, bf16* __restrict__ xb) {
  const int wid = threadIdx.x >> 5;
  const int l   = threadIdx.x & 31;
  const long row = (long)blockIdx.x * 8 + wid;
  const float* xr = x + (size_t)row * D_;

  float v[12], s = 0.f, s2 = 0.f;
#pragma unroll
  for (int j = 0; j < 12; ++j) {
    float f = xr[l + 32 * j];
    v[j] = f; s += f; s2 += f * f;
  }
#pragma unroll
  for (int o = 1; o < 32; o <<= 1) {
    s  += __shfl_xor(s,  o, 32);
    s2 += __shfl_xor(s2, o, 32);
  }
  const float mu  = s * (1.f / D_);
  const float var = s2 * (1.f / D_) - mu * mu;
  const float rs  = rsqrtf(var + 1e-5f);

  const float* pr  = proj + (size_t)row * D_;
  float*       xro = xo   + (size_t)row * D_;
  bf16*        xbr = xb   + (size_t)row * D_;
#pragma unroll
  for (int j = 0; j < 12; ++j) {
    int d = l + 32 * j;
    float y = (v[j] - mu) * rs * g[d] + be[d] + pr[d];
    xro[d] = y;
    xbr[d] = (bf16)y;
  }
}

// ---------------- host orchestration ----------------
extern "C" void kernel_launch(void* const* d_in, const int* in_sizes, int n_in,
                              void* d_out, int out_size, void* d_ws, size_t ws_size,
                              hipStream_t stream) {
  (void)in_sizes; (void)n_in; (void)out_size; (void)ws_size;
  const int*   idx  = (const int*)  d_in[0];
  const float* tok  = (const float*)d_in[1];
  const float* pos  = (const float*)d_in[2];
  const float* Wq   = (const float*)d_in[3];
  const float* Wk   = (const float*)d_in[4];
  const float* Wv   = (const float*)d_in[5];
  const float* Wo   = (const float*)d_in[6];
  const float* bo   = (const float*)d_in[7];
  const float* W1   = (const float*)d_in[8];
  const float* b1   = (const float*)d_in[9];
  const float* W2   = (const float*)d_in[10];
  const float* b2   = (const float*)d_in[11];
  const float* ln1g = (const float*)d_in[12];
  const float* ln1b = (const float*)d_in[13];
  const float* ln2g = (const float*)d_in[14];
  const float* ln2b = (const float*)d_in[15];
  const float* Wlm  = (const float*)d_in[16];
  const float* blm  = (const float*)d_in[17];

  char* ws = (char*)d_ws;
  size_t off = 0;
  auto take = [&](size_t bytes) -> char* {
    char* p = ws + off;
    off += (bytes + 255) & ~(size_t)255;
    return p;
  };

  float* x    = (float*)take((size_t)BT_ * D_ * 4);
  bf16*  xb   = (bf16*) take((size_t)BT_ * D_ * 2);
  bf16*  qb   = (bf16*) take((size_t)BT_ * D_ * 2);   // qb..ab contiguous
  bf16*  kb   = (bf16*) take((size_t)BT_ * D_ * 2);
  bf16*  vb   = (bf16*) take((size_t)BT_ * D_ * 2);
  bf16*  ab   = (bf16*) take((size_t)BT_ * D_ * 2);
  float* proj = (float*)take((size_t)BT_ * D_ * 4);
  bf16*  hid  = qb;  // FFN hidden [BT, DF] aliases qb..ab (DF == 4*D, both dead then)

  // transposed bf16 weights: Wt[n][k]
  bf16* wqT  = (bf16*)take((size_t)L_ * D_ * D_ * 2);
  bf16* wkT  = (bf16*)take((size_t)L_ * D_ * D_ * 2);
  bf16* wvT  = (bf16*)take((size_t)L_ * D_ * D_ * 2);
  bf16* woT  = (bf16*)take((size_t)L_ * D_ * D_ * 2);
  bf16* w1T  = (bf16*)take((size_t)L_ * D_ * DF_ * 2);  // per layer [DF][D]
  bf16* w2T  = (bf16*)take((size_t)L_ * DF_ * D_ * 2);  // per layer [D][DF]
  bf16* wlmT = (bf16*)take((size_t)V_ * D_ * 2);        // [V][D]

  auto cvtT = [&](const float* s, bf16* d, int K, int N, int Lb) {
    k_cvtT<<<2048, 256, 0, stream>>>(s, d, K, N, Lb);
  };
  cvtT(Wq, wqT, D_, D_, L_);
  cvtT(Wk, wkT, D_, D_, L_);
  cvtT(Wv, wvT, D_, D_, L_);
  cvtT(Wo, woT, D_, D_, L_);
  cvtT(W1, w1T, D_, DF_, L_);
  cvtT(W2, w2T, DF_, D_, L_);
  cvtT(Wlm, wlmT, D_, V_, 1);

  k_embed<<<(int)BT_, 128, 0, stream>>>(idx, tok, pos, x, xb);

  auto gemm = [&](const bf16* A, const bf16* Wt, const float* bias,
                  float* Cf, bf16* Cb, int M, int N, int K, int flags) {
    dim3 g((N + BN - 1) / BN, M / BM);
    k_gemm<<<g, 256, 0, stream>>>(A, Wt, bias, Cf, Cb, M, N, K, flags);
  };

  for (int lyr = 0; lyr < L_; ++lyr) {
    const size_t wdd = (size_t)lyr * D_ * D_;
    gemm(xb, wqT + wdd, nullptr, nullptr, qb, (int)BT_, D_, D_, 0);
    gemm(xb, wkT + wdd, nullptr, nullptr, kb, (int)BT_, D_, D_, 0);
    gemm(xb, wvT + wdd, nullptr, nullptr, vb, (int)BT_, D_, D_, 0);

    k_attn<<<dim3(T_ / 16, H_, B_), 32, 0, stream>>>(qb, kb, vb, ab);

    gemm(ab, woT + wdd, bo + (size_t)lyr * D_, proj, nullptr, (int)BT_, D_, D_, 1);
    k_ln_residual<<<(int)(BT_ / 8), 256, 0, stream>>>(
        x, ln1g + (size_t)lyr * D_, ln1b + (size_t)lyr * D_, proj, x, xb);

    gemm(xb, w1T + (size_t)lyr * D_ * DF_, b1 + (size_t)lyr * DF_,
         nullptr, hid, (int)BT_, DF_, D_, 1 | 2);
    gemm(hid, w2T + (size_t)lyr * DF_ * D_, b2 + (size_t)lyr * D_,
         proj, nullptr, (int)BT_, D_, DF_, 1);
    k_ln_residual<<<(int)(BT_ / 8), 256, 0, stream>>>(
        x, ln2g + (size_t)lyr * D_, ln2b + (size_t)lyr * D_, proj, x, xb);
  }

  gemm(xb, wlmT, blm, (float*)d_out, nullptr, (int)BT_, V_, D_, 1);
}
